// EfficientAttention_10127532884249
// MI455X (gfx1250) — matrix operand order repaired and checked by reading, not compiled, not run on hardware
//
#include <hip/hip_runtime.h>
#include <hip/hip_bf16.h>

typedef __bf16 bf16_t;
typedef __attribute__((ext_vector_type(16))) __bf16 v16bf;
typedef __attribute__((ext_vector_type(8)))  __bf16 v8bf;
typedef __attribute__((ext_vector_type(4)))  __bf16 v4bf;
typedef __attribute__((ext_vector_type(2)))  __bf16 v2bf;
typedef __attribute__((ext_vector_type(8)))  float  v8f;

#define NB     16
#define CH     256
#define SPA    4096
#define NHEADS 8
#define HCH    32

static __device__ __forceinline__ bf16_t f2bf(float x) { return (bf16_t)x; }

static __device__ __forceinline__ float4 ld4(const float* p) {
  return *(const float4*)p;
}

static __device__ __forceinline__ v8f wmma_bf16(v16bf a, v16bf b, v8f c) {
  return __builtin_amdgcn_wmma_f32_16x16x32_bf16(false, a, false, b, (short)0, c,
                                                 false, false);
}

// A-fragment K index for element e (16x32 bf16 A layout, ISA 7.12.2)
static __device__ __forceinline__ int ka_idx(int e, int half) {
  return e + half * 8 + (e >= 8 ? 8 : 0);
}

// ---------------------------------------------------------------------------
// Kernel 1: out[n][o][s] = sum_c W[o][c] * X[n][c][s] + bias[o]
// grid (SPA/64, CH/128, NB), block 256 (8 waves, 4x2). Wave tile: 32x32.
// Register-prefetch pipelined, float4 staging.
// ---------------------------------------------------------------------------
__global__ void __launch_bounds__(256)
ea_proj_gemm(const float* __restrict__ W, const float* __restrict__ bias,
             const float* __restrict__ X, float* __restrict__ out) {
  __shared__ bf16_t Asm[128][32];  // [m][k]
  __shared__ bf16_t Bsm[64][32];   // [s][k] (transposed: K contiguous)
  const int n  = blockIdx.z;
  const int m0 = blockIdx.y * 128;
  const int s0 = blockIdx.x * 64;
  const int t = threadIdx.x;
  const int wave = t >> 5, lane = t & 31;
  const int half = lane >> 4, l16 = lane & 15;
  const int wm = wave & 3;   // M group (32 rows)
  const int wn = wave >> 2;  // N group (32 cols)
  const float* Xn = X + (size_t)n * CH * SPA;

  // staging coords
  const int arow = t >> 3;          // +0/32/64/96 per pass, 32 rows/pass
  const int akk  = (t & 7) * 4;     // 8 lanes cover one 128B row
  const int bkk2 = (t >> 4) * 2;    // k pair
  const int bss  = (t & 15) * 4;    // s quad

  float4 apre[4], bpre0, bpre1;
#pragma unroll
  for (int p = 0; p < 4; ++p)
    apre[p] = ld4(&W[(size_t)(m0 + arow + 32 * p) * CH + akk]);
  bpre0 = ld4(&Xn[(size_t)bkk2 * SPA + s0 + bss]);
  bpre1 = ld4(&Xn[(size_t)(bkk2 + 1) * SPA + s0 + bss]);

  v8f acc[2][2] = {};
  for (int k0 = 0; k0 < CH; k0 += 32) {
    // commit prefetched slab to LDS
#pragma unroll
    for (int p = 0; p < 4; ++p) {
      v4bf av = { f2bf(apre[p].x), f2bf(apre[p].y), f2bf(apre[p].z), f2bf(apre[p].w) };
      *(v4bf*)&Asm[arow + 32 * p][akk] = av;
    }
    {
      float4 r0 = bpre0, r1 = bpre1;
      v2bf p0 = { f2bf(r0.x), f2bf(r1.x) };
      v2bf p1 = { f2bf(r0.y), f2bf(r1.y) };
      v2bf p2 = { f2bf(r0.z), f2bf(r1.z) };
      v2bf p3 = { f2bf(r0.w), f2bf(r1.w) };
      *(v2bf*)&Bsm[bss + 0][bkk2] = p0;
      *(v2bf*)&Bsm[bss + 1][bkk2] = p1;
      *(v2bf*)&Bsm[bss + 2][bkk2] = p2;
      *(v2bf*)&Bsm[bss + 3][bkk2] = p3;
    }
    __syncthreads();
    // issue next slab's global loads; they complete under the WMMAs
    if (k0 + 32 < CH) {
      int kn = k0 + 32;
#pragma unroll
      for (int p = 0; p < 4; ++p)
        apre[p] = ld4(&W[(size_t)(m0 + arow + 32 * p) * CH + kn + akk]);
      bpre0 = ld4(&Xn[(size_t)(kn + bkk2) * SPA + s0 + bss]);
      bpre1 = ld4(&Xn[(size_t)(kn + bkk2 + 1) * SPA + s0 + bss]);
    }
    // fragments + 4 WMMAs
    v16bf a0, a1, b0, b1;
#pragma unroll
    for (int e = 0; e < 16; ++e) {
      int kA = ka_idx(e, half);
      a0[e] = Asm[wm * 32 + l16][kA];
      a1[e] = Asm[wm * 32 + 16 + l16][kA];
      int kB = ka_idx(e, half);
      b0[e] = Bsm[wn * 32 + l16][kB];
      b1[e] = Bsm[wn * 32 + 16 + l16][kB];
    }
    acc[0][0] = wmma_bf16(a0, b0, acc[0][0]);
    acc[0][1] = wmma_bf16(a0, b1, acc[0][1]);
    acc[1][0] = wmma_bf16(a1, b0, acc[1][0]);
    acc[1][1] = wmma_bf16(a1, b1, acc[1][1]);
    __syncthreads();
  }
  float* On = out + (size_t)n * CH * SPA;
#pragma unroll
  for (int mi = 0; mi < 2; ++mi)
#pragma unroll
    for (int ni = 0; ni < 2; ++ni)
#pragma unroll
      for (int r = 0; r < 8; ++r) {
        int m = m0 + wm * 32 + mi * 16 + r + half * 8;
        int s = s0 + wn * 32 + ni * 16 + l16;
        On[(size_t)m * SPA + s] = acc[mi][ni][r] + bias[m];
      }
}

// ---------------------------------------------------------------------------
// Kernel 2: in-place softmax over s per row of k. grid NB*CH, block 256.
// ---------------------------------------------------------------------------
__global__ void __launch_bounds__(256) ea_k_softmax(float* __restrict__ kbuf) {
  float* p = kbuf + (size_t)blockIdx.x * SPA;
  const int t = threadIdx.x;
  __shared__ float red[256];
  float4 loc[4];
#pragma unroll
  for (int i = 0; i < 4; ++i) loc[i] = ld4(&p[(t + i * 256) * 4]);
  float m = -3.4e38f;
#pragma unroll
  for (int i = 0; i < 4; ++i)
    m = fmaxf(m, fmaxf(fmaxf(loc[i].x, loc[i].y), fmaxf(loc[i].z, loc[i].w)));
  red[t] = m; __syncthreads();
  for (int w = 128; w > 0; w >>= 1) {
    if (t < w) red[t] = fmaxf(red[t], red[t + w]);
    __syncthreads();
  }
  m = red[0]; __syncthreads();
  float s = 0.f;
#pragma unroll
  for (int i = 0; i < 4; ++i) {
    loc[i].x = __expf(loc[i].x - m); loc[i].y = __expf(loc[i].y - m);
    loc[i].z = __expf(loc[i].z - m); loc[i].w = __expf(loc[i].w - m);
    s += loc[i].x + loc[i].y + loc[i].z + loc[i].w;
  }
  red[t] = s; __syncthreads();
  for (int w = 128; w > 0; w >>= 1) {
    if (t < w) red[t] += red[t + w];
    __syncthreads();
  }
  float inv = 1.f / red[0];
#pragma unroll
  for (int i = 0; i < 4; ++i) {
    float4 o = { loc[i].x * inv, loc[i].y * inv, loc[i].z * inv, loc[i].w * inv };
    *(float4*)&p[(t + i * 256) * 4] = o;
  }
}

// ---------------------------------------------------------------------------
// Kernel 3: ctx[n][h][ck][cv] = sum_s ksm[n][h*32+ck][s] * v[n][h*32+cv][s]
// grid (NHEADS, NB), block 256; 8 waves split K=4096, LDS atomic reduce.
// ---------------------------------------------------------------------------
__global__ void __launch_bounds__(256)
ea_ctx_gemm(const float* __restrict__ ksm, const float* __restrict__ v,
            float* __restrict__ ctx) {
  const int h = blockIdx.x, n = blockIdx.y;
  const int t = threadIdx.x, wave = t >> 5, lane = t & 31;
  const int half = lane >> 4, l16 = lane & 15;
  const float* Ar0 = ksm + ((size_t)n * CH + h * HCH + l16) * SPA;
  const float* Ar1 = Ar0 + (size_t)16 * SPA;
  const float* Br0 = v + ((size_t)n * CH + h * HCH + l16) * SPA;
  const float* Br1 = Br0 + (size_t)16 * SPA;
  v8f acc[2][2] = {};
  for (int k0 = wave * 512; k0 < wave * 512 + 512; k0 += 32) {
    const int ab = k0 + half * 16;   // A: run  [ab..ab+15]
    const int bb = k0 + half * 16;   // B: run  [bb..bb+15]
    float4 qa[4], qc[4], qb[4], qd[4];
    qa[0] = ld4(Ar0 + ab);      qa[1] = ld4(Ar0 + ab + 4);
    qa[2] = ld4(Ar0 + ab + 8);  qa[3] = ld4(Ar0 + ab + 12);
    qc[0] = ld4(Ar1 + ab);      qc[1] = ld4(Ar1 + ab + 4);
    qc[2] = ld4(Ar1 + ab + 8);  qc[3] = ld4(Ar1 + ab + 12);
    qb[0] = ld4(Br0 + bb);      qb[1] = ld4(Br0 + bb + 4);
    qb[2] = ld4(Br0 + bb + 8);  qb[3] = ld4(Br0 + bb + 12);
    qd[0] = ld4(Br1 + bb);      qd[1] = ld4(Br1 + bb + 4);
    qd[2] = ld4(Br1 + bb + 8);  qd[3] = ld4(Br1 + bb + 12);
    v16bf a0, a1, b0, b1;
#pragma unroll
    for (int q = 0; q < 4; ++q) {
      const float* fa = (const float*)&qa[q];
      const float* fc = (const float*)&qc[q];
      const float* fb = (const float*)&qb[q];
      const float* fd = (const float*)&qd[q];
#pragma unroll
      for (int j = 0; j < 4; ++j) {
        a0[q * 4 + j] = f2bf(fa[j]);
        a1[q * 4 + j] = f2bf(fc[j]);
        b0[q * 4 + j] = f2bf(fb[j]);
        b1[q * 4 + j] = f2bf(fd[j]);
      }
    }
    acc[0][0] = wmma_bf16(a0, b0, acc[0][0]);
    acc[0][1] = wmma_bf16(a0, b1, acc[0][1]);
    acc[1][0] = wmma_bf16(a1, b0, acc[1][0]);
    acc[1][1] = wmma_bf16(a1, b1, acc[1][1]);
  }
  __shared__ float cred[HCH][HCH];
  for (int i = t; i < HCH * HCH; i += 256) ((float*)cred)[i] = 0.f;
  __syncthreads();
#pragma unroll
  for (int mi = 0; mi < 2; ++mi)
#pragma unroll
    for (int ni = 0; ni < 2; ++ni)
#pragma unroll
      for (int r = 0; r < 8; ++r)
        atomicAdd(&cred[mi * 16 + r + half * 8][ni * 16 + l16], acc[mi][ni][r]);
  __syncthreads();
  float* oc = ctx + ((size_t)n * NHEADS + h) * HCH * HCH;
  for (int i = t; i < HCH * HCH; i += 256) oc[i] = ((float*)cred)[i];
}

// ---------------------------------------------------------------------------
// Kernel 4 (fused): q channel-softmax -> att = ctx^T @ q_sm (per head, WMMA)
//                   -> out = Wr @ att + br + X.   grid (SPA/64, NB), block 256.
// ---------------------------------------------------------------------------
__global__ void __launch_bounds__(256)
ea_fused_out(const float* __restrict__ qbuf, const float* __restrict__ ctx,
             const float* __restrict__ Wr, const float* __restrict__ br,
             const float* __restrict__ X, float* __restrict__ out) {
  __shared__ bf16_t q_lds[64][CH];    // [s][c]   (B layout for att GEMM)
  __shared__ bf16_t att_lds[64][CH];  // [s][vch] (B layout for final GEMM)
  __shared__ bf16_t wr_lds[CH][32];   // [o][k]   staged per K-step
  const int n  = blockIdx.y;
  const int s0 = blockIdx.x * 64;
  const int t = threadIdx.x;
  const int wave = t >> 5, lane = t & 31;
  const int half = lane >> 4, l16 = lane & 15;

  // --- q channel softmax (coalesced along s per channel) ---
  for (int cidx = t; cidx < NHEADS * 64; cidx += 256) {
    int h = cidx >> 6, ss = cidx & 63;
    const float* qp = qbuf + ((size_t)n * CH + h * HCH) * SPA + (s0 + ss);
    float vals[HCH];
    float m = -3.4e38f;
#pragma unroll
    for (int ck = 0; ck < HCH; ++ck) {
      vals[ck] = qp[(size_t)ck * SPA];
      m = fmaxf(m, vals[ck]);
    }
    float s = 0.f;
#pragma unroll
    for (int ck = 0; ck < HCH; ++ck) { vals[ck] = __expf(vals[ck] - m); s += vals[ck]; }
    float inv = 1.f / s;
#pragma unroll
    for (int ck = 0; ck < HCH; ++ck) q_lds[ss][h * HCH + ck] = f2bf(vals[ck] * inv);
  }
  __syncthreads();

  // --- per-head att = ctx^T (cv x ck) @ q_sm (ck x 64); wave == head ---
  {
    const float* cp = ctx + ((size_t)n * NHEADS + wave) * HCH * HCH;
    v16bf a0, a1;
#pragma unroll
    for (int e = 0; e < 16; ++e) {
      int kA = ka_idx(e, half);  // ck
      a0[e] = f2bf(cp[kA * HCH + l16]);
      a1[e] = f2bf(cp[kA * HCH + l16 + 16]);
    }
#pragma unroll
    for (int ns = 0; ns < 4; ++ns) {
      v16bf b;
#pragma unroll
      for (int e = 0; e < 16; ++e)
        b[e] = q_lds[ns * 16 + l16][wave * HCH + ka_idx(e, half)];
      v8f c0 = {}, c1 = {};
      c0 = wmma_bf16(a0, b, c0);
      c1 = wmma_bf16(a1, b, c1);
#pragma unroll
      for (int r = 0; r < 8; ++r) {
        att_lds[ns * 16 + l16][wave * HCH + r + half * 8]      = f2bf(c0[r]);
        att_lds[ns * 16 + l16][wave * HCH + 16 + r + half * 8] = f2bf(c1[r]);
      }
    }
  }
  __syncthreads();

  // --- out = Wr(256x256) @ att(256x64) + br + X ---
  const int wrow = t >> 3;        // +32 per pass
  const int wkk  = (t & 7) * 4;   // 8 lanes cover a 128B row
  v8f acc[2][4] = {};
  for (int k0 = 0; k0 < CH; k0 += 32) {
#pragma unroll
    for (int p = 0; p < 8; ++p) {
      float4 wv = ld4(&Wr[(size_t)(wrow + 32 * p) * CH + k0 + wkk]);
      v4bf pk = { f2bf(wv.x), f2bf(wv.y), f2bf(wv.z), f2bf(wv.w) };
      *(v4bf*)&wr_lds[wrow + 32 * p][wkk] = pk;
    }
    __syncthreads();
    v16bf b[4];
#pragma unroll
    for (int ns = 0; ns < 4; ++ns)
#pragma unroll
      for (int e = 0; e < 16; ++e)
        b[ns][e] = att_lds[ns * 16 + l16][k0 + ka_idx(e, half)];
#pragma unroll
    for (int mi = 0; mi < 2; ++mi) {
      v16bf a;
#pragma unroll
      for (int e = 0; e < 16; ++e)
        a[e] = wr_lds[wave * 32 + mi * 16 + l16][ka_idx(e, half)];
#pragma unroll
      for (int ns = 0; ns < 4; ++ns) acc[mi][ns] = wmma_bf16(a, b[ns], acc[mi][ns]);
    }
    __syncthreads();
  }
#pragma unroll
  for (int mi = 0; mi < 2; ++mi)
#pragma unroll
    for (int ns = 0; ns < 4; ++ns)
#pragma unroll
      for (int r = 0; r < 8; ++r) {
        int o = wave * 32 + mi * 16 + r + half * 8;
        int s = s0 + ns * 16 + l16;
        size_t idx = ((size_t)n * CH + o) * SPA + s;
        out[idx] = acc[mi][ns][r] + br[o] + X[idx];
      }
}

// ---------------------------------------------------------------------------
extern "C" void kernel_launch(void* const* d_in, const int* in_sizes, int n_in,
                              void* d_out, int out_size, void* d_ws, size_t ws_size,
                              hipStream_t stream) {
  const float* X  = (const float*)d_in[0];
  const float* Wk = (const float*)d_in[1];
  const float* bk = (const float*)d_in[2];
  const float* Wq = (const float*)d_in[3];
  const float* bq = (const float*)d_in[4];
  const float* Wv = (const float*)d_in[5];
  const float* bv = (const float*)d_in[6];
  const float* Wr = (const float*)d_in[7];
  const float* br = (const float*)d_in[8];
  float* out = (float*)d_out;

  const size_t proj_bytes = (size_t)NB * CH * SPA * sizeof(float);  // 64 MB
  float* kbuf = (float*)((char*)d_ws + 0);
  float* vbuf = (float*)((char*)d_ws + proj_bytes);
  float* qbuf = (float*)((char*)d_ws + 2 * proj_bytes);
  float* cbuf = (float*)((char*)d_ws + 3 * proj_bytes);  // 512 KB ctx

  dim3 gp(SPA / 64, CH / 128, NB);
  ea_proj_gemm<<<gp, 256, 0, stream>>>(Wk, bk, X, kbuf);
  ea_proj_gemm<<<gp, 256, 0, stream>>>(Wv, bv, X, vbuf);
  ea_proj_gemm<<<gp, 256, 0, stream>>>(Wq, bq, X, qbuf);
  ea_k_softmax<<<NB * CH, 256, 0, stream>>>(kbuf);
  ea_ctx_gemm<<<dim3(NHEADS, NB), 256, 0, stream>>>(kbuf, vbuf, cbuf);
  ea_fused_out<<<dim3(SPA / 64, NB), 256, 0, stream>>>(qbuf, cbuf, Wr, br, X, out);
}
